// Mask_Rcnn_14276471292179
// MI455X (gfx1250) — compile-verified, matrix operand order checked by reading, not run on hardware
//
#include <hip/hip_runtime.h>
#include <math.h>

typedef __bf16 bf16;
typedef __attribute__((ext_vector_type(16))) __bf16 v16bf;
typedef __attribute__((ext_vector_type(8)))  __bf16 v8bf;
typedef __attribute__((ext_vector_type(8)))  float  v8f;

#define NEGV (-1e30f)

// ---------------------------------------------------------------------------
// Fragment helpers
// ---------------------------------------------------------------------------
__device__ __forceinline__ v16bf cvt16(const float* v) {
    v16bf r;
#pragma unroll
    for (int i = 0; i < 16; ++i) r[i] = (bf16)v[i];
    return r;
}

__device__ __forceinline__ v16bf zero16bf() {
    v16bf z;
#pragma unroll
    for (int i = 0; i < 16; ++i) z[i] = (bf16)0.f;
    return z;
}

// B fragment: row bn of bf16 [N][K]; lane needs K runs [koff,koff+8) and
// [koff+16,koff+24) -> two 16-byte vector loads. K must be a multiple of 32.
__device__ __forceinline__ v16bf load_bt(const bf16* __restrict__ wt, int bn, int N,
                                         int K, int koff) {
    if (bn < N) {
        const v8bf* p = (const v8bf*)(wt + (size_t)bn * K + koff);
        v8bf lo = p[0];
        v8bf hi = p[2];               // +16 elements
        return __builtin_shufflevector(lo, hi, 0, 1, 2, 3, 4, 5, 6, 7,
                                       8, 9, 10, 11, 12, 13, 14, 15);
    }
    return zero16bf();
}

// A fragment from row-major fp32 A[M][K]: four float4 loads + pack to bf16.
__device__ __forceinline__ v16bf load_at(const float* __restrict__ A, int am, int M,
                                         int K, int koff) {
    float v[16];
    if (am < M) {
        const float4* p = (const float4*)(A + (size_t)am * K + koff);
        float4 q0 = p[0], q1 = p[1], q2 = p[4], q3 = p[5];
        v[0] = q0.x; v[1] = q0.y; v[2] = q0.z; v[3] = q0.w;
        v[4] = q1.x; v[5] = q1.y; v[6] = q1.z; v[7] = q1.w;
        v[8] = q2.x; v[9] = q2.y; v[10] = q2.z; v[11] = q2.w;
        v[12] = q3.x; v[13] = q3.y; v[14] = q3.z; v[15] = q3.w;
    } else {
#pragma unroll
        for (int i = 0; i < 16; ++i) v[i] = 0.f;
    }
    return cvt16(v);
}

// ---------------------------------------------------------------------------
// Elementwise / conversion kernels
// ---------------------------------------------------------------------------
__global__ void f2bf(const float* __restrict__ s, bf16* __restrict__ d, int n) {
    int i = blockIdx.x * blockDim.x + threadIdx.x;
    if (i < n) d[i] = (bf16)s[i];
}

// src is [K,N] row-major fp32, dst is [N,K] row-major bf16 (transposed)
__global__ void f2bf_t(const float* __restrict__ s, bf16* __restrict__ d, int K, int N) {
    int i = blockIdx.x * blockDim.x + threadIdx.x;
    if (i >= N * K) return;
    int n = i / K, k = i - n * K;
    d[i] = (bf16)s[(size_t)k * N + n];
}

// out[c,y,x] = lat[c,y,x] + top[c,y/2,x/2]   (top is H/2 x W/2). In-place safe.
__global__ void upsample_add(const float* __restrict__ lat, const float* __restrict__ top,
                             float* __restrict__ out, int C, int H, int W) {
    int i = blockIdx.x * blockDim.x + threadIdx.x;
    if (i >= C * H * W) return;
    int x = i % W, y = (i / W) % H, c = i / (W * H);
    out[i] = lat[i] + top[((size_t)c * (H >> 1) + (y >> 1)) * (W >> 1) + (x >> 1)];
}

// out[c,y,x] = in[c,2y,2x]
__global__ void subsample2(const float* __restrict__ in, float* __restrict__ out,
                           int C, int Hi, int Wi) {
    int Ho = Hi >> 1, Wo = Wi >> 1;
    int i = blockIdx.x * blockDim.x + threadIdx.x;
    if (i >= C * Ho * Wo) return;
    int x = i % Wo, y = (i / Wo) % Ho, c = i / (Wo * Ho);
    out[i] = in[((size_t)c * Hi + 2 * y) * Wi + 2 * x];
}

// ---------------------------------------------------------------------------
// Implicit-GEMM conv via WMMA (bf16 x bf16 -> f32), specialized on kernel size.
//   in : [batch][Cin][H][W] fp32       wt : [Cout][Cin*KSZ*KSZ] bf16 (OIHW)
//   out: [batch][Cout][H][W] fp32      SAME padding, stride 1
// One 16(M) x 32(N) super-tile per wave (A fragment reused across 2 B tiles),
// 8 waves / block, grid.y = batch. Cin*KSZ*KSZ must be a multiple of 32.
// ---------------------------------------------------------------------------
template <int KSZ>
__global__ void conv_wmma(const float* __restrict__ in, const bf16* __restrict__ wt,
                          const float* __restrict__ bias, float* __restrict__ out,
                          int Cin, int H, int W, int Cout, int relu, int tiles) {
    int wave = threadIdx.x >> 5;
    int lane = threadIdx.x & 31;
    int tile = blockIdx.x * 8 + wave;
    if (tile >= tiles) return;                 // wave-uniform: EXEC stays all-1s
    int M = H * W;
    in  += (size_t)blockIdx.y * Cin  * M;
    out += (size_t)blockIdx.y * Cout * M;
    int tilesN = (Cout + 31) >> 5;             // 32-wide N super-tiles
    int n0 = (tile % tilesN) << 5;
    int m0 = (tile / tilesN) << 4;
    int half = lane >> 4, l15 = lane & 15;
    int am = m0 + l15;
    int ay = am / W, ax = am - ay * W;
    const int kk2 = KSZ * KSZ, kh = KSZ >> 1;
    int Ktot = Cin * kk2;
    int bn0 = n0 + l15, bn1 = n0 + 16 + l15;
    bool amv = am < M;
    v8f acc0 = {0.f, 0.f, 0.f, 0.f, 0.f, 0.f, 0.f, 0.f};
    v8f acc1 = {0.f, 0.f, 0.f, 0.f, 0.f, 0.f, 0.f, 0.f};
    for (int k0 = 0; k0 < Ktot; k0 += 32) {
        if (bn0 < Cout) __builtin_prefetch(wt + (size_t)bn0 * Ktot + k0 + 64, 0, 1);
        float av[16];
#pragma unroll
        for (int i = 0; i < 8; ++i) {
            int kb = k0 + ((i & 4) << 2) + (half << 3) + ((i & 3) << 1);
#pragma unroll
            for (int t = 0; t < 2; ++t) {
                int kk = kb + t;
                float v = 0.f;
                if (amv) {
                    int c  = kk / kk2;                      // constant divisor
                    int rs = kk - c * kk2;
                    int r  = rs / KSZ - kh;
                    int s  = rs - (rs / KSZ) * KSZ - kh;
                    int yy = ay + r, xx = ax + s;
                    if (KSZ == 1 ||
                        ((unsigned)yy < (unsigned)H && (unsigned)xx < (unsigned)W))
                        v = in[((size_t)c * H + yy) * W + xx];
                }
                av[2 * i + t] = v;
            }
        }
        v16bf a  = cvt16(av);
        int koff = k0 + (half << 3);
        v16bf b0 = load_bt(wt, bn0, Cout, Ktot, koff);
        v16bf b1 = load_bt(wt, bn1, Cout, Ktot, koff);
        acc0 = __builtin_amdgcn_wmma_f32_16x16x32_bf16(false, a, false, b0,
                                                       (short)0, acc0, false, false);
        acc1 = __builtin_amdgcn_wmma_f32_16x16x32_bf16(false, a, false, b1,
                                                       (short)0, acc1, false, false);
    }
#pragma unroll
    for (int nt = 0; nt < 2; ++nt) {
        int nc = n0 + (nt << 4) + l15;         // C/D layout: N = lane&15
        if (nc < Cout) {
            float bv = bias ? bias[nc] : 0.f;
            v8f acc = nt ? acc1 : acc0;
#pragma unroll
            for (int v = 0; v < 8; ++v) {
                int mm = m0 + v + (half << 3); // M = vgpr + 8*(lane>=16)
                if (mm < M) {
                    float r = acc[v] + bv;
                    out[(size_t)nc * M + mm] = relu ? fmaxf(r, 0.f) : r;
                }
            }
        }
    }
}

// ---------------------------------------------------------------------------
// Row-major GEMM via WMMA: C[M,N] = act(A[M,K] * Bt[N,K]^T + bias)
// A fp32 row-major (vector-loaded, packed to bf16), Bt bf16 [N][K].
// 16(M) x 32(N) per wave. K must be a multiple of 32.
// ---------------------------------------------------------------------------
__global__ void gemm_wmma(const float* __restrict__ A, const bf16* __restrict__ Bt,
                          const float* __restrict__ bias, float* __restrict__ C,
                          int M, int N, int K, int relu, int tiles) {
    int wave = threadIdx.x >> 5;
    int lane = threadIdx.x & 31;
    int tile = blockIdx.x * 8 + wave;
    if (tile >= tiles) return;
    int tilesN = (N + 31) >> 5;
    int n0 = (tile % tilesN) << 5;
    int m0 = (tile / tilesN) << 4;
    int half = lane >> 4, l15 = lane & 15;
    int am = m0 + l15;
    int bn0 = n0 + l15, bn1 = n0 + 16 + l15;
    v8f acc0 = {0.f, 0.f, 0.f, 0.f, 0.f, 0.f, 0.f, 0.f};
    v8f acc1 = {0.f, 0.f, 0.f, 0.f, 0.f, 0.f, 0.f, 0.f};
    for (int k0 = 0; k0 < K; k0 += 32) {
        if (bn0 < N) __builtin_prefetch(Bt + (size_t)bn0 * K + k0 + 64, 0, 1);
        int koff = k0 + (half << 3);
        v16bf a  = load_at(A, am, M, K, koff);
        v16bf b0 = load_bt(Bt, bn0, N, K, koff);
        v16bf b1 = load_bt(Bt, bn1, N, K, koff);
        acc0 = __builtin_amdgcn_wmma_f32_16x16x32_bf16(false, a, false, b0,
                                                       (short)0, acc0, false, false);
        acc1 = __builtin_amdgcn_wmma_f32_16x16x32_bf16(false, a, false, b1,
                                                       (short)0, acc1, false, false);
    }
#pragma unroll
    for (int nt = 0; nt < 2; ++nt) {
        int nc = n0 + (nt << 4) + l15;
        if (nc < N) {
            float bv = bias ? bias[nc] : 0.f;
            v8f acc = nt ? acc1 : acc0;
#pragma unroll
            for (int v = 0; v < 8; ++v) {
                int mm = m0 + v + (half << 3);
                if (mm < M) {
                    float r = acc[v] + bv;
                    C[(size_t)mm * N + nc] = relu ? fmaxf(r, 0.f) : r;
                }
            }
        }
    }
}

// ---------------------------------------------------------------------------
// RPN post: anchors + softmax score + box decode + clip + min-size filter
// cls: [6][H][W], loc: [12][H][W]. Writes boxes[offset+idx], scores[offset+idx].
// ---------------------------------------------------------------------------
__global__ void rpn_post(const float* __restrict__ cls, const float* __restrict__ loc,
                         int H, int W, int stride, float scale,
                         float* __restrict__ boxes, float* __restrict__ scores, int offset) {
    int idx = blockIdx.x * blockDim.x + threadIdx.x;
    int total = H * W * 3;
    if (idx >= total) return;
    int a = idx % 3, pix = idx / 3;
    int x = pix % W, y = pix / W;
    int HW = H * W;
    float rr = (a == 0) ? 0.5f : (a == 1 ? 1.f : 2.f);
    float sq = sqrtf(rr);
    float ha = scale * sq, wa = scale / sq;
    float cy = (y + 0.5f) * stride, cx = (x + 0.5f) * stride;
    float l0 = cls[(size_t)(2 * a) * HW + pix];
    float l1 = cls[(size_t)(2 * a + 1) * HW + pix];
    float sc = 1.f / (1.f + expf(l0 - l1));
    float d0 = loc[(size_t)(4 * a + 0) * HW + pix];
    float d1 = loc[(size_t)(4 * a + 1) * HW + pix];
    float d2 = loc[(size_t)(4 * a + 2) * HW + pix];
    float d3 = loc[(size_t)(4 * a + 3) * HW + pix];
    float pcx = d0 * wa + cx, pcy = d1 * ha + cy;
    float pw = expf(fminf(d2, 4.f)) * wa, ph = expf(fminf(d3, 4.f)) * ha;
    float bx1 = fminf(fmaxf(pcx - 0.5f * pw, 0.f), 768.f);
    float by1 = fminf(fmaxf(pcy - 0.5f * ph, 0.f), 512.f);
    float bx2 = fminf(fmaxf(pcx + 0.5f * pw, 0.f), 768.f);
    float by2 = fminf(fmaxf(pcy + 0.5f * ph, 0.f), 512.f);
    if ((bx2 - bx1) < 16.f || (by2 - by1) < 16.f) sc = NEGV;
    int o = offset + idx;
    boxes[(size_t)o * 4 + 0] = bx1; boxes[(size_t)o * 4 + 1] = by1;
    boxes[(size_t)o * 4 + 2] = bx2; boxes[(size_t)o * 4 + 3] = by2;
    scores[o] = sc;
}

// ---------------------------------------------------------------------------
// Iterative top-K selection (single block). Mutates s. Lowest-index tie-break
// to match jnp.argmax.
// ---------------------------------------------------------------------------
__global__ void topk_kernel(float* __restrict__ s, int n, int K,
                            int* __restrict__ oi, float* __restrict__ os) {
    __shared__ float sm[1024];
    __shared__ int   si[1024];
    int tid = threadIdx.x, bs = blockDim.x;
    for (int it = 0; it < K; ++it) {
        float best = -INFINITY; int bi = 0;
        for (int j = tid; j < n; j += bs) {
            float v = s[j];
            if (v > best) { best = v; bi = j; }
        }
        sm[tid] = best; si[tid] = bi;
        __syncthreads();
        for (int st = bs >> 1; st > 0; st >>= 1) {
            if (tid < st) {
                if (sm[tid + st] > sm[tid] ||
                    (sm[tid + st] == sm[tid] && si[tid + st] < si[tid])) {
                    sm[tid] = sm[tid + st]; si[tid] = si[tid + st];
                }
            }
            __syncthreads();
        }
        if (tid == 0) { oi[it] = si[0]; os[it] = sm[0]; s[si[0]] = -INFINITY; }
        __syncthreads();
    }
}

// ---------------------------------------------------------------------------
// Greedy fixed-iteration NMS. blockIdx.x = class (or 0). Mutates s.
// ks[it] = selected score (pre-suppression) if valid else NEG  -> "kept".
// ---------------------------------------------------------------------------
__global__ void nms_kernel(const float* __restrict__ boxes, float* __restrict__ s,
                           const float* __restrict__ area, int n, int K, float th,
                           int* __restrict__ ki, int* __restrict__ kv, float* __restrict__ ks) {
    int c = blockIdx.x;
    boxes += (size_t)c * n * 4; s += (size_t)c * n; area += (size_t)c * n;
    ki += (size_t)c * K; kv += (size_t)c * K; ks += (size_t)c * K;
    __shared__ float sm[256];
    __shared__ int   si[256];
    int tid = threadIdx.x, bs = blockDim.x;
    for (int it = 0; it < K; ++it) {
        float best = -INFINITY; int bi = 0;
        for (int j = tid; j < n; j += bs) {
            float v = s[j];
            if (v > best) { best = v; bi = j; }
        }
        sm[tid] = best; si[tid] = bi;
        __syncthreads();
        for (int st = bs >> 1; st > 0; st >>= 1) {
            if (tid < st) {
                if (sm[tid + st] > sm[tid] ||
                    (sm[tid + st] == sm[tid] && si[tid + st] < si[tid])) {
                    sm[tid] = sm[tid + st]; si[tid] = si[tid + st];
                }
            }
            __syncthreads();
        }
        int j = si[0]; float sj = sm[0];
        __syncthreads();
        int valid = (sj > NEGV * 0.5f) ? 1 : 0;
        if (tid == 0) { ki[it] = j; kv[it] = valid; ks[it] = valid ? sj : NEGV; }
        float bx1 = boxes[j * 4 + 0], by1 = boxes[j * 4 + 1];
        float bx2 = boxes[j * 4 + 2], by2 = boxes[j * 4 + 3];
        float aj = area[j];
        for (int q = tid; q < n; q += bs) {
            float ix1 = fmaxf(bx1, boxes[q * 4 + 0]);
            float iy1 = fmaxf(by1, boxes[q * 4 + 1]);
            float ix2 = fminf(bx2, boxes[q * 4 + 2]);
            float iy2 = fminf(by2, boxes[q * 4 + 3]);
            float inter = fmaxf(ix2 - ix1, 0.f) * fmaxf(iy2 - iy1, 0.f);
            float iou = inter / fmaxf(aj + area[q] - inter, 1e-6f);
            if (iou >= th) s[q] = NEGV;
        }
        __syncthreads();
    }
}

__global__ void areas_kernel(const float* __restrict__ b, float* __restrict__ area, int n) {
    int i = blockIdx.x * blockDim.x + threadIdx.x;
    if (i >= n) return;
    area[i] = fmaxf(b[i * 4 + 2] - b[i * 4 + 0], 0.f) * fmaxf(b[i * 4 + 3] - b[i * 4 + 1], 0.f);
}

__global__ void gather_boxes(const float* __restrict__ boxes, const int* __restrict__ idxs,
                             float* __restrict__ out, int n) {
    int i = blockIdx.x * blockDim.x + threadIdx.x;
    if (i >= n) return;
    int j = idxs[i];
    out[i * 4 + 0] = boxes[(size_t)j * 4 + 0];
    out[i * 4 + 1] = boxes[(size_t)j * 4 + 1];
    out[i * 4 + 2] = boxes[(size_t)j * 4 + 2];
    out[i * 4 + 3] = boxes[(size_t)j * 4 + 3];
}

__global__ void roi_build(const int* __restrict__ ki, const int* __restrict__ kv,
                          const float* __restrict__ cand, float* __restrict__ roi,
                          int* __restrict__ roi_valid, int K) {
    int i = blockIdx.x * blockDim.x + threadIdx.x;
    if (i >= K) return;
    int v = kv[i]; roi_valid[i] = v;
    int j = ki[i];
    if (v) {
        roi[i * 4 + 0] = cand[j * 4 + 0]; roi[i * 4 + 1] = cand[j * 4 + 1];
        roi[i * 4 + 2] = cand[j * 4 + 2]; roi[i * 4 + 3] = cand[j * 4 + 3];
    } else {
        roi[i * 4 + 0] = 0.f; roi[i * 4 + 1] = 0.f; roi[i * 4 + 2] = 1.f; roi[i * 4 + 3] = 1.f;
    }
}

// ---------------------------------------------------------------------------
// Multi-scale ROIAlign, sampling_ratio=2. Level dims hardcoded (strides 4..32).
// out[r][c][py][px], c in [0,256)
// ---------------------------------------------------------------------------
__global__ void roi_align_ms(const float* __restrict__ f0, const float* __restrict__ f1,
                             const float* __restrict__ f2, const float* __restrict__ f3,
                             const float* __restrict__ rois, int R, int P,
                             float* __restrict__ out) {
    int idx = blockIdx.x * blockDim.x + threadIdx.x;
    int total = R * 256 * P * P;
    if (idx >= total) return;
    int px = idx % P, py = (idx / P) % P;
    int c = (idx / (P * P)) % 256;
    int r = idx / (P * P * 256);
    float x1 = rois[r * 4 + 0], y1 = rois[r * 4 + 1];
    float x2 = rois[r * 4 + 2], y2 = rois[r * 4 + 3];
    float bw = fmaxf(x2 - x1, 1e-6f), bh = fmaxf(y2 - y1, 1e-6f);
    int lvl = (int)floorf(4.f + log2f(sqrtf(bw * bh) / 224.f + 1e-8f));
    lvl = lvl < 2 ? 2 : (lvl > 5 ? 5 : lvl);
    lvl -= 2;
    const float* f = (lvl == 0) ? f0 : (lvl == 1) ? f1 : (lvl == 2) ? f2 : f3;
    int H = 128 >> lvl, W = 192 >> lvl;
    float scale = 1.f / (float)(4 << lvl);
    float rx1 = x1 * scale, ry1 = y1 * scale;
    float rw = fmaxf((x2 - x1) * scale, 1.f);
    float rh = fmaxf((y2 - y1) * scale, 1.f);
    const float* fc = f + (size_t)c * H * W;
    float inv = 1.f / (float)(2 * P);
    float acc = 0.f;
#pragma unroll
    for (int sy = 0; sy < 2; ++sy)
#pragma unroll
        for (int sx = 0; sx < 2; ++sx) {
            float gy = ry1 + (2 * py + sy + 0.5f) * rh * inv;
            float gx = rx1 + (2 * px + sx + 0.5f) * rw * inv;
            gy = fminf(fmaxf(gy, 0.f), (float)(H - 1));
            gx = fminf(fmaxf(gx, 0.f), (float)(W - 1));
            int y0 = (int)floorf(gy), x0 = (int)floorf(gx);
            int y1i = (y0 + 1 < H - 1) ? y0 + 1 : H - 1;
            int x1i = (x0 + 1 < W - 1) ? x0 + 1 : W - 1;
            float ly = gy - y0, lx = gx - x0, hy = 1.f - ly, hx = 1.f - lx;
            acc += fc[y0 * W + x0] * hy * hx + fc[y0 * W + x1i] * hy * lx
                 + fc[y1i * W + x0] * ly * hx + fc[y1i * W + x1i] * ly * lx;
        }
    out[idx] = acc * 0.25f;
}

__global__ void loc_scale(float* __restrict__ loc, int n) {
    int i = blockIdx.x * blockDim.x + threadIdx.x;
    if (i >= n) return;
    loc[i] *= ((i & 3) < 2) ? 0.1f : 0.2f;
}

__global__ void softmax81(const float* __restrict__ logits, float* __restrict__ probs, int R) {
    int r = blockIdx.x * blockDim.x + threadIdx.x;
    if (r >= R) return;
    const float* l = logits + (size_t)r * 81;
    float mx = -INFINITY;
    for (int j = 0; j < 81; ++j) mx = fmaxf(mx, l[j]);
    float sum = 0.f;
    float* p = probs + (size_t)r * 81;
    for (int j = 0; j < 81; ++j) { float e = expf(l[j] - mx); p[j] = e; sum += e; }
    float inv = 1.f / sum;
    for (int j = 0; j < 81; ++j) p[j] *= inv;
}

// Per-class decode+clip+threshold: boxes_c[(cls-1)*R + r], sc likewise.
__global__ void class_decode(const float* __restrict__ roi, const int* __restrict__ roi_valid,
                             const float* __restrict__ loc, const float* __restrict__ probs,
                             float* __restrict__ boxes_c, float* __restrict__ sc, int R) {
    int idx = blockIdx.x * blockDim.x + threadIdx.x;
    if (idx >= 80 * R) return;
    int cls = idx / R + 1;
    int r = idx - (cls - 1) * R;
    float x1 = roi[r * 4 + 0], y1 = roi[r * 4 + 1];
    float x2 = roi[r * 4 + 2], y2 = roi[r * 4 + 3];
    float w = x2 - x1, h = y2 - y1;
    float cx = x1 + 0.5f * w, cy = y1 + 0.5f * h;
    const float* d = loc + (size_t)r * 324 + cls * 4;
    float pcx = d[0] * w + cx, pcy = d[1] * h + cy;
    float pw = expf(fminf(d[2], 4.f)) * w, ph = expf(fminf(d[3], 4.f)) * h;
    float bx1 = fminf(fmaxf(pcx - 0.5f * pw, 0.f), 768.f);
    float by1 = fminf(fmaxf(pcy - 0.5f * ph, 0.f), 512.f);
    float bx2 = fminf(fmaxf(pcx + 0.5f * pw, 0.f), 768.f);
    float by2 = fminf(fmaxf(pcy + 0.5f * ph, 0.f), 512.f);
    boxes_c[(size_t)idx * 4 + 0] = bx1; boxes_c[(size_t)idx * 4 + 1] = by1;
    boxes_c[(size_t)idx * 4 + 2] = bx2; boxes_c[(size_t)idx * 4 + 3] = by2;
    float p = probs[(size_t)r * 81 + cls];
    if (!roi_valid[r] || p < 0.05f) p = NEGV;
    sc[idx] = p;
}

__global__ void det_build(const int* __restrict__ det_i, const float* __restrict__ det_s,
                          const int* __restrict__ ki_c, const float* __restrict__ boxes_c,
                          float* __restrict__ pre, float* __restrict__ det_boxes) {
    int i = threadIdx.x;
    if (i >= 100) return;
    int di = det_i[i];
    int cls = di / 100, slot = di - cls * 100;
    int bidx = ki_c[cls * 100 + slot];
    const float* b = boxes_c + ((size_t)cls * 1000 + bidx) * 4;
    pre[i * 6 + 0] = b[0]; pre[i * 6 + 1] = b[1];
    pre[i * 6 + 2] = b[2]; pre[i * 6 + 3] = b[3];
    pre[i * 6 + 4] = det_s[i];
    pre[i * 6 + 5] = (float)cls;
    det_boxes[i * 4 + 0] = b[0]; det_boxes[i * 4 + 1] = b[1];
    det_boxes[i * 4 + 2] = b[2]; det_boxes[i * 4 + 3] = b[3];
}

// ConvTranspose2d(256,256,2,stride=2) + bias + relu
// x: [100,256,14,14], w: [c][d][a][b], out: [100,256,28,28]
__global__ void deconv2x2(const float* __restrict__ x, const float* __restrict__ w,
                          const float* __restrict__ bias, float* __restrict__ out) {
    int idx = blockIdx.x * blockDim.x + threadIdx.x;
    if (idx >= 100 * 256 * 28 * 28) return;
    int ox = idx % 28, oy = (idx / 28) % 28;
    int d = (idx / 784) % 256, n = idx / (784 * 256);
    int i = oy >> 1, a = oy & 1, j = ox >> 1, b2 = ox & 1;
    const float* xp = x + (size_t)n * 256 * 196 + i * 14 + j;
    const float* wp = w + d * 4 + a * 2 + b2;
    float acc = 0.f;
    for (int c = 0; c < 256; ++c)
        acc += xp[(size_t)c * 196] * wp[(size_t)c * 1024];
    out[idx] = fmaxf(acc + bias[d], 0.f);
}

// select class mask channel + sigmoid
__global__ void mask_select(const float* __restrict__ pre, const float* __restrict__ mlog,
                            float* __restrict__ out) {
    int idx = blockIdx.x * blockDim.x + threadIdx.x;
    if (idx >= 100 * 784) return;
    int r = idx / 784, p = idx - r * 784;
    int cls = (int)pre[r * 6 + 5] + 1;
    float v = mlog[((size_t)r * 81 + cls) * 784 + p];
    out[idx] = 1.f / (1.f + expf(-v));
}

// ---------------------------------------------------------------------------
// Host orchestration
// ---------------------------------------------------------------------------
extern "C" void kernel_launch(void* const* d_in, const int* in_sizes, int n_in,
                              void* d_out, int out_size, void* d_ws, size_t ws_size,
                              hipStream_t stream) {
    static const int LH[5] = {128, 64, 32, 16, 8};
    static const int LW[5] = {192, 96, 48, 24, 12};
    static const int LS[5] = {4, 8, 16, 32, 64};
    static const float LSC[5] = {32.f, 64.f, 128.f, 256.f, 512.f};
    static const int CIN[4] = {256, 512, 1024, 2048};
    const int NA = 98208;

    const float* c_in[4];
    for (int i = 0; i < 4; ++i) c_in[i] = (const float*)d_in[i];
    const float *lat_w[4], *lat_b[4], *fpn_w[4], *fpn_b[4];
    for (int i = 0; i < 4; ++i) {
        lat_w[i] = (const float*)d_in[4 + 4 * i];
        lat_b[i] = (const float*)d_in[5 + 4 * i];
        fpn_w[i] = (const float*)d_in[6 + 4 * i];
        fpn_b[i] = (const float*)d_in[7 + 4 * i];
    }
    const float* rpn_w     = (const float*)d_in[20];
    const float* rpn_b     = (const float*)d_in[21];
    const float* rpn_cls_w = (const float*)d_in[22];
    const float* rpn_cls_b = (const float*)d_in[23];
    const float* rpn_loc_w = (const float*)d_in[24];
    const float* rpn_loc_b = (const float*)d_in[25];
    const float* fast_w1   = (const float*)d_in[26];
    const float* fast_b1   = (const float*)d_in[27];
    const float* fast_w2   = (const float*)d_in[28];
    const float* fast_b2   = (const float*)d_in[29];
    const float* cls_w     = (const float*)d_in[30];
    const float* cls_b     = (const float*)d_in[31];
    const float* loc_w     = (const float*)d_in[32];
    const float* loc_b     = (const float*)d_in[33];
    const float *mask_w[4], *mask_b[4];
    for (int i = 0; i < 4; ++i) {
        mask_w[i] = (const float*)d_in[34 + 2 * i];
        mask_b[i] = (const float*)d_in[35 + 2 * i];
    }
    const float* mask_dw = (const float*)d_in[42];
    const float* mask_db = (const float*)d_in[43];
    const float* mask_pw = (const float*)d_in[44];
    const float* mask_pb = (const float*)d_in[45];

    // bump allocator on d_ws
    size_t off = 0;
    auto alloc = [&](size_t bytes) -> void* {
        off = (off + 255) & ~(size_t)255;
        void* p = (char*)d_ws + off;
        off += bytes;
        return p;
    };
    auto aF = [&](size_t n) { return (float*)alloc(n * sizeof(float)); };
    auto aB = [&](size_t n) { return (bf16*)alloc(n * sizeof(bf16)); };
    auto aI = [&](size_t n) { return (int*)alloc(n * sizeof(int)); };
    auto blk = [](size_t n) { return dim3((unsigned)((n + 255) / 256)); };

    auto cvt = [&](const float* s, bf16* d, size_t n) {
        f2bf<<<blk(n), 256, 0, stream>>>(s, d, (int)n);
    };
    auto conv = [&](const float* in, const bf16* w, const float* bias, float* out,
                    int Cin, int H, int W, int Cout, int k, int relu, int batch) {
        int tiles = ((H * W + 15) / 16) * ((Cout + 31) / 32);
        dim3 g((tiles + 7) / 8, batch);
        if (k == 1)
            conv_wmma<1><<<g, 256, 0, stream>>>(in, w, bias, out, Cin, H, W, Cout, relu, tiles);
        else
            conv_wmma<3><<<g, 256, 0, stream>>>(in, w, bias, out, Cin, H, W, Cout, relu, tiles);
    };
    auto gemm = [&](const float* A, const bf16* Bt, const float* bias, float* C,
                    int M, int N, int K, int relu) {
        int tiles = ((M + 15) / 16) * ((N + 31) / 32);
        gemm_wmma<<<(tiles + 7) / 8, 256, 0, stream>>>(A, Bt, bias, C, M, N, K, relu, tiles);
    };

    // ---- weight conversion to bf16 ([N][K] layouts) ----
    bf16 *lat_wb[4], *fpn_wb[4], *mask_wb[4];
    for (int i = 0; i < 4; ++i) {
        lat_wb[i] = aB((size_t)256 * CIN[i]);  cvt(lat_w[i], lat_wb[i], (size_t)256 * CIN[i]);
        fpn_wb[i] = aB(589824);                cvt(fpn_w[i], fpn_wb[i], 589824);
        mask_wb[i] = aB(589824);               cvt(mask_w[i], mask_wb[i], 589824);
    }
    bf16* rpn_wb = aB(589824);      cvt(rpn_w, rpn_wb, 589824);
    bf16* rpn_cls_wb = aB(1536);    cvt(rpn_cls_w, rpn_cls_wb, 1536);
    bf16* rpn_loc_wb = aB(3072);    cvt(rpn_loc_w, rpn_loc_wb, 3072);
    bf16* mask_pwb = aB(20736);     cvt(mask_pw, mask_pwb, 20736);
    bf16* w1t = aB((size_t)1024 * 12544);
    f2bf_t<<<blk((size_t)1024 * 12544), 256, 0, stream>>>(fast_w1, w1t, 12544, 1024);
    bf16* w2t = aB((size_t)1024 * 1024);
    f2bf_t<<<blk((size_t)1024 * 1024), 256, 0, stream>>>(fast_w2, w2t, 1024, 1024);
    bf16* clst = aB((size_t)81 * 1024);
    f2bf_t<<<blk((size_t)81 * 1024), 256, 0, stream>>>(cls_w, clst, 1024, 81);
    bf16* loct = aB((size_t)324 * 1024);
    f2bf_t<<<blk((size_t)324 * 1024), 256, 0, stream>>>(loc_w, loct, 1024, 324);

    // ---- FPN ----
    float* top[4];
    float* feat[5];
    for (int i = 0; i < 4; ++i) top[i] = aF((size_t)256 * LH[i] * LW[i]);
    for (int i = 0; i < 4; ++i) feat[i] = aF((size_t)256 * LH[i] * LW[i]);
    feat[4] = aF((size_t)256 * LH[4] * LW[4]);

    conv(c_in[3], lat_wb[3], lat_b[3], top[3], CIN[3], LH[3], LW[3], 256, 1, 0, 1);
    for (int i = 2; i >= 0; --i) {
        conv(c_in[i], lat_wb[i], lat_b[i], top[i], CIN[i], LH[i], LW[i], 256, 1, 0, 1);
        size_t n = (size_t)256 * LH[i] * LW[i];
        upsample_add<<<blk(n), 256, 0, stream>>>(top[i], top[i + 1], top[i], 256, LH[i], LW[i]);
    }
    for (int i = 0; i < 4; ++i)
        conv(top[i], fpn_wb[i], fpn_b[i], feat[i], 256, LH[i], LW[i], 256, 3, 0, 1);
    subsample2<<<blk((size_t)256 * LH[4] * LW[4]), 256, 0, stream>>>(feat[3], feat[4], 256, LH[3], LW[3]);

    // ---- RPN heads + proposals ----
    float* rpn_hid = aF((size_t)256 * LH[0] * LW[0]);
    float* rpn_cls = aF((size_t)6 * LH[0] * LW[0]);
    float* rpn_loc = aF((size_t)12 * LH[0] * LW[0]);
    float* boxes_all = aF((size_t)NA * 4);
    float* scores_all = aF(NA);
    int aoff = 0;
    for (int l = 0; l < 5; ++l) {
        int H = LH[l], W = LW[l];
        conv(feat[l], rpn_wb, rpn_b, rpn_hid, 256, H, W, 256, 3, 1, 1);
        conv(rpn_hid, rpn_cls_wb, rpn_cls_b, rpn_cls, 256, H, W, 6, 1, 0, 1);
        conv(rpn_hid, rpn_loc_wb, rpn_loc_b, rpn_loc, 256, H, W, 12, 1, 0, 1);
        rpn_post<<<blk((size_t)H * W * 3), 256, 0, stream>>>(
            rpn_cls, rpn_loc, H, W, LS[l], LSC[l], boxes_all, scores_all, aoff);
        aoff += H * W * 3;
    }

    int* top_i = aI(6000);
    float* top_s = aF(6000);
    topk_kernel<<<1, 1024, 0, stream>>>(scores_all, NA, 6000, top_i, top_s);
    float* cand = aF((size_t)6000 * 4);
    gather_boxes<<<blk(6000), 256, 0, stream>>>(boxes_all, top_i, cand, 6000);
    float* area1 = aF(6000);
    areas_kernel<<<blk(6000), 256, 0, stream>>>(cand, area1, 6000);
    int* ki = aI(1000); int* kv = aI(1000); float* ksc = aF(1000);
    nms_kernel<<<1, 256, 0, stream>>>(cand, top_s, area1, 6000, 1000, 0.7f, ki, kv, ksc);
    float* roi = aF(4000); int* roi_valid = aI(1000);
    roi_build<<<blk(1000), 256, 0, stream>>>(ki, kv, cand, roi, roi_valid, 1000);

    // ---- Fast R-CNN head ----
    float* x7 = aF((size_t)1000 * 256 * 49);
    roi_align_ms<<<blk((size_t)1000 * 256 * 49), 256, 0, stream>>>(
        feat[0], feat[1], feat[2], feat[3], roi, 1000, 7, x7);
    float* h1 = aF((size_t)1000 * 1024);
    float* h2 = aF((size_t)1000 * 1024);
    float* flog = aF((size_t)1000 * 81);
    float* floc = aF((size_t)1000 * 324);
    float* probs = aF((size_t)1000 * 81);
    gemm(x7, w1t, fast_b1, h1, 1000, 1024, 12544, 1);
    gemm(h1, w2t, fast_b2, h2, 1000, 1024, 1024, 1);
    gemm(h2, clst, cls_b, flog, 1000, 81, 1024, 0);
    gemm(h2, loct, loc_b, floc, 1000, 324, 1024, 0);
    loc_scale<<<blk((size_t)1000 * 324), 256, 0, stream>>>(floc, 1000 * 324);
    softmax81<<<blk(1000), 256, 0, stream>>>(flog, probs, 1000);

    // ---- per-class decode + NMS + global top-100 ----
    float* boxes_c = aF((size_t)80 * 1000 * 4);
    float* scc = aF((size_t)80 * 1000);
    class_decode<<<blk((size_t)80 * 1000), 256, 0, stream>>>(
        roi, roi_valid, floc, probs, boxes_c, scc, 1000);
    float* area2 = aF((size_t)80 * 1000);
    areas_kernel<<<blk((size_t)80 * 1000), 256, 0, stream>>>(boxes_c, area2, 80 * 1000);
    int* ki_c = aI(8000); int* kv_c = aI(8000); float* kept = aF(8000);
    nms_kernel<<<80, 256, 0, stream>>>(boxes_c, scc, area2, 1000, 100, 0.5f, ki_c, kv_c, kept);
    int* det_i = aI(100); float* det_s = aF(100);
    topk_kernel<<<1, 1024, 0, stream>>>(kept, 8000, 100, det_i, det_s);
    float* det_boxes = aF(400);
    det_build<<<1, 128, 0, stream>>>(det_i, det_s, ki_c, boxes_c, (float*)d_out, det_boxes);

    // ---- mask head ----
    float* mA = aF((size_t)100 * 256 * 196);
    float* mB = aF((size_t)100 * 256 * 196);
    roi_align_ms<<<blk((size_t)100 * 256 * 196), 256, 0, stream>>>(
        feat[0], feat[1], feat[2], feat[3], det_boxes, 100, 14, mA);
    float* src = mA; float* dst = mB;
    for (int i = 0; i < 4; ++i) {
        conv(src, mask_wb[i], mask_b[i], dst, 256, 14, 14, 256, 3, 1, 100);
        float* t = src; src = dst; dst = t;
    }
    float* mup = aF((size_t)100 * 256 * 784);
    deconv2x2<<<blk((size_t)100 * 256 * 784), 256, 0, stream>>>(src, mask_dw, mask_db, mup);
    float* mlog = aF((size_t)100 * 81 * 784);
    conv(mup, mask_pwb, mask_pb, mlog, 256, 28, 28, 81, 1, 0, 100);
    mask_select<<<blk((size_t)100 * 784), 256, 0, stream>>>(
        (const float*)d_out, mlog, (float*)d_out + 600);
}
